// CosineSimilarity_23072564314299
// MI455X (gfx1250) — compile-verified, matrix operand order checked by reading, not run on hardware
//
#include <hip/hip_runtime.h>
#include <stdint.h>

// ---------------------------------------------------------------------------
// CosineSimilarity (8192x1024) x (8192x1024) -> (8192x8192), scaled by 1/0.05
// Split-bf16 (hi/lo) WMMA GEMM, gfx1250.
//  Fast path (ws >= 64MB): pre-normalized bf16 hi/lo panels in workspace,
//    GEMM hot loop = async global->LDS + ds_load_b128 + v_wmma only.
//  Fallback path: fused fp32->bf16 conversion inside GEMM (round-1 scheme).
// ---------------------------------------------------------------------------

typedef __attribute__((ext_vector_type(16))) __bf16 v16bf;
typedef __attribute__((ext_vector_type(8)))  __bf16 v8bf;
typedef __attribute__((ext_vector_type(4)))  __bf16 v4bf;
typedef __attribute__((ext_vector_type(8)))  float  v8f;

#define N_ROWS 8192
#define DIM    1024
#define INV_TEMP 20.0f

// ===========================================================================
// FAST PATH
// ===========================================================================

// ---- kernel A: fused row-norm + normalize + bf16 hi/lo split --------------
__global__ void __launch_bounds__(256)
csim_norm_convert_kernel(const float* __restrict__ x1,
                         const float* __restrict__ x2,
                         __bf16* __restrict__ Ah, __bf16* __restrict__ Al,
                         __bf16* __restrict__ Bh, __bf16* __restrict__ Bl) {
  const int row = blockIdx.x;                         // 0 .. 2*N_ROWS-1
  const bool isA = row < N_ROWS;
  const int r = isA ? row : row - N_ROWS;
  const float* x = (isA ? x1 : x2) + (size_t)r * DIM;
  __bf16* hiDst = (isA ? Ah : Bh) + (size_t)r * DIM;
  __bf16* loDst = (isA ? Al : Bl) + (size_t)r * DIM;

  float4 v = ((const float4*)x)[threadIdx.x];         // 256*4 = 1024 elems
  float s = v.x * v.x + v.y * v.y + v.z * v.z + v.w * v.w;
  #pragma unroll
  for (int off = 16; off > 0; off >>= 1)
    s += __shfl_xor(s, off, 32);

  __shared__ float part[8];
  __shared__ float sInv;
  if ((threadIdx.x & 31) == 0) part[threadIdx.x >> 5] = s;
  __syncthreads();
  if (threadIdx.x == 0) {
    float t = 0.f;
    #pragma unroll
    for (int i = 0; i < 8; ++i) t += part[i];
    sInv = 1.0f / fmaxf(sqrtf(t), 1e-8f);
  }
  __syncthreads();
  const float inv = sInv;

  float y[4] = {v.x * inv, v.y * inv, v.z * inv, v.w * inv};
  v4bf hv, lv;
  #pragma unroll
  for (int i = 0; i < 4; ++i) {
    __bf16 h = (__bf16)y[i];
    hv[i] = h;
    lv[i] = (__bf16)(y[i] - (float)h);
  }
  ((v4bf*)hiDst)[threadIdx.x] = hv;
  ((v4bf*)loDst)[threadIdx.x] = lv;
}

// ---- kernel B: WMMA GEMM on pre-split bf16 panels -------------------------
// Block tile 128x128, K chunk = 32. 8 waves as 2(M) x 4(N); each wave 64x32.
// LDS per buffer: 4 panels (Ah, Al, Bh, Bl), 128 rows x 64B, stride 80B
// (conflict-free). Double buffered: 2 * 4 * 10240 = 80 KB.

#define BK2    32
#define ROWB2  80                    // 32 bf16 (64B) + 16B pad
#define PANB2  (128 * ROWB2)         // 10240 B per panel
#define BUFB2  (4 * PANB2)           // 40960 B per buffer

__device__ __forceinline__ void async_load_panel_bf16(
    const __bf16* gbase, int rowBase, int kbase, unsigned ldsBase,
    int wid, int lane) {
  const int rlane = lane >> 2;       // 0..7 : row within 8-row group
  const int c16   = lane & 3;        // 0..3 : which 16B of the 64B row
  unsigned long long sbase = (unsigned long long)(uintptr_t)gbase;
  #pragma unroll
  for (int j = 0; j < 2; ++j) {
    int row = j * 64 + wid * 8 + rlane;               // 0..127 across 8 waves
    unsigned goff  = (unsigned)(((rowBase + row) * DIM + kbase) * 2 + c16 * 16);
    unsigned laddr = ldsBase + (unsigned)(row * ROWB2 + c16 * 16);
    asm volatile("global_load_async_to_lds_b128 %0, %1, %2"
                 :: "v"(laddr), "v"(goff), "s"(sbase)
                 : "memory");
  }
}

__device__ __forceinline__ v16bf load_frag_bf16(const unsigned char* p) {
  v8bf a = *(const v8bf*)p;          // K 0..7   (or 8..15 for lanes 16-31)
  v8bf b = *(const v8bf*)(p + 32);   // K 16..23 (or 24..31)
  v16bf r;
  #pragma unroll
  for (int i = 0; i < 8; ++i) { r[i] = a[i]; r[i + 8] = b[i]; }
  return r;
}

__global__ void __launch_bounds__(256)
csim_gemm_pre_kernel(const __bf16* __restrict__ Ah, const __bf16* __restrict__ Al,
                     const __bf16* __restrict__ Bh, const __bf16* __restrict__ Bl,
                     float* __restrict__ out) {
  __shared__ __attribute__((aligned(16))) unsigned char smem[2 * BUFB2];

  const int tid    = threadIdx.x;
  const int wid    = tid >> 5;
  const int lane   = tid & 31;
  const int laneLo = lane & 15;
  const int laneHi = lane >> 4;
  const int waveM  = wid >> 2;       // 0..1
  const int waveN  = wid & 3;        // 0..3
  const int bM = blockIdx.y * 128;
  const int bN = blockIdx.x * 128;

  v8f acc[4][2];
  #pragma unroll
  for (int m = 0; m < 4; ++m)
    #pragma unroll
    for (int n = 0; n < 2; ++n)
      acc[m][n] = (v8f)(0.0f);

  const unsigned sbase = (unsigned)(uintptr_t)smem;

  // Prologue: chunk 0 -> buffer 0 (panels: Ah, Al, Bh, Bl)
  async_load_panel_bf16(Ah, bM, 0, sbase + 0 * PANB2, wid, lane);
  async_load_panel_bf16(Al, bM, 0, sbase + 1 * PANB2, wid, lane);
  async_load_panel_bf16(Bh, bN, 0, sbase + 2 * PANB2, wid, lane);
  async_load_panel_bf16(Bl, bN, 0, sbase + 3 * PANB2, wid, lane);

  const int KCH = DIM / BK2;         // 32 chunks
  for (int kc = 0; kc < KCH; ++kc) {
    const int cur = kc & 1;
    if (kc + 1 < KCH) {
      const int nxt = cur ^ 1;
      const unsigned nb = sbase + nxt * BUFB2;
      const int kb = (kc + 1) * BK2;
      async_load_panel_bf16(Ah, bM, kb, nb + 0 * PANB2, wid, lane);
      async_load_panel_bf16(Al, bM, kb, nb + 1 * PANB2, wid, lane);
      async_load_panel_bf16(Bh, bN, kb, nb + 2 * PANB2, wid, lane);
      async_load_panel_bf16(Bl, bN, kb, nb + 3 * PANB2, wid, lane);
      // 8 newer ops outstanding; <=8 left => chunk kc (this wave) landed
      asm volatile("s_wait_asynccnt 0x8" ::: "memory");
    } else {
      asm volatile("s_wait_asynccnt 0x0" ::: "memory");
    }
    __syncthreads();                 // chunk kc visible to all waves

    const unsigned char* buf = smem + cur * BUFB2;
    const unsigned char* pAh = buf + 0 * PANB2;
    const unsigned char* pAl = buf + 1 * PANB2;
    const unsigned char* pBh = buf + 2 * PANB2;
    const unsigned char* pBl = buf + 3 * PANB2;
    const unsigned fo = (unsigned)(laneHi * 16);   // K-group select per lane

    v16bf Bhi[2], Blo[2];
    #pragma unroll
    for (int n = 0; n < 2; ++n) {
      int r = waveN * 32 + n * 16 + laneLo;
      Bhi[n] = load_frag_bf16(pBh + r * ROWB2 + fo);
      Blo[n] = load_frag_bf16(pBl + r * ROWB2 + fo);
    }
    #pragma unroll
    for (int m = 0; m < 4; ++m) {
      int r = waveM * 64 + m * 16 + laneLo;
      v16bf Ahi = load_frag_bf16(pAh + r * ROWB2 + fo);
      v16bf Alo = load_frag_bf16(pAl + r * ROWB2 + fo);
      #pragma unroll
      for (int n = 0; n < 2; ++n) {
        acc[m][n] = __builtin_amdgcn_wmma_f32_16x16x32_bf16(
            false, Ahi, false, Bhi[n], (short)0, acc[m][n], false, false);
        acc[m][n] = __builtin_amdgcn_wmma_f32_16x16x32_bf16(
            false, Ahi, false, Blo[n], (short)0, acc[m][n], false, false);
        acc[m][n] = __builtin_amdgcn_wmma_f32_16x16x32_bf16(
            false, Alo, false, Bhi[n], (short)0, acc[m][n], false, false);
      }
    }
    __syncthreads();                 // nobody overwrites buf[cur] early
  }

  #pragma unroll
  for (int m = 0; m < 4; ++m) {
    #pragma unroll
    for (int n = 0; n < 2; ++n) {
      int col = bN + waveN * 32 + n * 16 + laneLo;
      #pragma unroll
      for (int v = 0; v < 8; ++v) {
        int row = bM + waveM * 64 + m * 16 + v + laneHi * 8;
        out[(size_t)row * N_ROWS + col] = acc[m][n][v] * INV_TEMP;
      }
    }
  }
}

// ===========================================================================
// FALLBACK PATH (ws too small): round-1 fused-conversion GEMM
// ===========================================================================

__global__ void __launch_bounds__(256)
csim_norms_kernel(const float* __restrict__ x1, const float* __restrict__ x2,
                  float* __restrict__ inv) {
  const int row = blockIdx.x;
  const float* x = (row < N_ROWS) ? (x1 + (size_t)row * DIM)
                                  : (x2 + (size_t)(row - N_ROWS) * DIM);
  float4 v = ((const float4*)x)[threadIdx.x];
  float s = v.x * v.x + v.y * v.y + v.z * v.z + v.w * v.w;
  #pragma unroll
  for (int off = 16; off > 0; off >>= 1)
    s += __shfl_xor(s, off, 32);
  __shared__ float part[8];
  if ((threadIdx.x & 31) == 0) part[threadIdx.x >> 5] = s;
  __syncthreads();
  if (threadIdx.x == 0) {
    float t = 0.f;
    #pragma unroll
    for (int i = 0; i < 8; ++i) t += part[i];
    inv[row] = 1.0f / fmaxf(sqrtf(t), 1e-8f);
  }
}

#define BK      32
#define ROWB    144
#define TILEB   (128 * ROWB)

__device__ __forceinline__ void async_load_chunk_f32(
    const float* gbase, int rowBase, int kbase, unsigned ldsBase,
    int wid, int lane) {
  const int rlane = lane >> 3;
  const int c16   = lane & 7;
  unsigned long long sbase = (unsigned long long)(uintptr_t)gbase;
  #pragma unroll
  for (int j = 0; j < 4; ++j) {
    int row = j * 32 + wid * 4 + rlane;
    unsigned goff  = (unsigned)(((rowBase + row) * DIM + kbase) * 4 + c16 * 16);
    unsigned laddr = ldsBase + (unsigned)(row * ROWB + c16 * 16);
    asm volatile("global_load_async_to_lds_b128 %0, %1, %2"
                 :: "v"(laddr), "v"(goff), "s"(sbase)
                 : "memory");
  }
}

__device__ __forceinline__ void load_frag_f32(const unsigned char* rowPtr,
                                              float scale, v16bf& hi, v16bf& lo) {
  const float4* p = (const float4*)rowPtr;
  float4 a = p[0];
  float4 b = p[1];
  const float4* q = (const float4*)(rowPtr + 64);
  float4 c = q[0];
  float4 d = q[1];
  float f[16] = {a.x, a.y, a.z, a.w, b.x, b.y, b.z, b.w,
                 c.x, c.y, c.z, c.w, d.x, d.y, d.z, d.w};
  #pragma unroll
  for (int i = 0; i < 16; ++i) {
    float y = f[i] * scale;
    __bf16 h = (__bf16)y;
    hi[i] = h;
    lo[i] = (__bf16)(y - (float)h);
  }
}

__global__ void __launch_bounds__(256)
csim_gemm_fused_kernel(const float* __restrict__ x1, const float* __restrict__ x2,
                       const float* __restrict__ inv, float* __restrict__ out) {
  __shared__ __attribute__((aligned(16))) unsigned char smem[4 * TILEB];

  const int tid    = threadIdx.x;
  const int wid    = tid >> 5;
  const int lane   = tid & 31;
  const int laneLo = lane & 15;
  const int laneHi = lane >> 4;
  const int waveM  = wid >> 2;
  const int waveN  = wid & 3;
  const int bM = blockIdx.y * 128;
  const int bN = blockIdx.x * 128;

  float invA[4], invB[2];
  #pragma unroll
  for (int m = 0; m < 4; ++m)
    invA[m] = inv[bM + waveM * 64 + m * 16 + laneLo];
  #pragma unroll
  for (int n = 0; n < 2; ++n)
    invB[n] = inv[N_ROWS + bN + waveN * 32 + n * 16 + laneLo];

  v8f acc[4][2];
  #pragma unroll
  for (int m = 0; m < 4; ++m)
    #pragma unroll
    for (int n = 0; n < 2; ++n)
      acc[m][n] = (v8f)(0.0f);

  const unsigned sbase = (unsigned)(uintptr_t)smem;
  async_load_chunk_f32(x1, bM, 0, sbase + 0,     wid, lane);
  async_load_chunk_f32(x2, bN, 0, sbase + TILEB, wid, lane);

  const int KCH = DIM / BK;
  for (int kc = 0; kc < KCH; ++kc) {
    const int cur = kc & 1;
    if (kc + 1 < KCH) {
      const int nxt = cur ^ 1;
      async_load_chunk_f32(x1, bM, (kc + 1) * BK, sbase + nxt * 2 * TILEB,         wid, lane);
      async_load_chunk_f32(x2, bN, (kc + 1) * BK, sbase + nxt * 2 * TILEB + TILEB, wid, lane);
      asm volatile("s_wait_asynccnt 0x8" ::: "memory");
    } else {
      asm volatile("s_wait_asynccnt 0x0" ::: "memory");
    }
    __syncthreads();

    const unsigned char* Ab = smem + cur * 2 * TILEB;
    const unsigned char* Bb = Ab + TILEB;

    v16bf Bhi[2], Blo[2];
    #pragma unroll
    for (int n = 0; n < 2; ++n) {
      int r = waveN * 32 + n * 16 + laneLo;
      load_frag_f32(Bb + r * ROWB + laneHi * 32, invB[n], Bhi[n], Blo[n]);
    }
    #pragma unroll
    for (int m = 0; m < 4; ++m) {
      int r = waveM * 64 + m * 16 + laneLo;
      v16bf Ahi, Alo;
      load_frag_f32(Ab + r * ROWB + laneHi * 32, invA[m], Ahi, Alo);
      #pragma unroll
      for (int n = 0; n < 2; ++n) {
        acc[m][n] = __builtin_amdgcn_wmma_f32_16x16x32_bf16(
            false, Ahi, false, Bhi[n], (short)0, acc[m][n], false, false);
        acc[m][n] = __builtin_amdgcn_wmma_f32_16x16x32_bf16(
            false, Ahi, false, Blo[n], (short)0, acc[m][n], false, false);
        acc[m][n] = __builtin_amdgcn_wmma_f32_16x16x32_bf16(
            false, Alo, false, Bhi[n], (short)0, acc[m][n], false, false);
      }
    }
    __syncthreads();
  }

  #pragma unroll
  for (int m = 0; m < 4; ++m) {
    #pragma unroll
    for (int n = 0; n < 2; ++n) {
      int col = bN + waveN * 32 + n * 16 + laneLo;
      #pragma unroll
      for (int v = 0; v < 8; ++v) {
        int row = bM + waveM * 64 + m * 16 + v + laneHi * 8;
        out[(size_t)row * N_ROWS + col] = acc[m][n][v] * INV_TEMP;
      }
    }
  }
}

// ------------------------------- launcher -----------------------------------
extern "C" void kernel_launch(void* const* d_in, const int* in_sizes, int n_in,
                              void* d_out, int out_size, void* d_ws, size_t ws_size,
                              hipStream_t stream) {
  const float* x1 = (const float*)d_in[0];
  const float* x2 = (const float*)d_in[1];
  float* out = (float*)d_out;

  const size_t PAN = (size_t)N_ROWS * DIM;          // elements per panel
  const size_t need = 4 * PAN * sizeof(__bf16);     // 64 MB

  if (ws_size >= need) {
    __bf16* Ah = (__bf16*)d_ws;
    __bf16* Al = Ah + PAN;
    __bf16* Bh = Al + PAN;
    __bf16* Bl = Bh + PAN;
    csim_norm_convert_kernel<<<2 * N_ROWS, 256, 0, stream>>>(x1, x2, Ah, Al, Bh, Bl);
    csim_gemm_pre_kernel<<<dim3(N_ROWS / 128, N_ROWS / 128), 256, 0, stream>>>(
        Ah, Al, Bh, Bl, out);
  } else {
    float* inv = (float*)d_ws;                      // 64 KB
    csim_norms_kernel<<<2 * N_ROWS, 256, 0, stream>>>(x1, x2, inv);
    csim_gemm_fused_kernel<<<dim3(N_ROWS / 128, N_ROWS / 128), 256, 0, stream>>>(
        x1, x2, inv, out);
  }
}
